// FullQuantumModel_43954695307725
// MI455X (gfx1250) — matrix operand order repaired and checked by reading, not compiled, not run on hardware
//
#include <hip/hip_runtime.h>
#include <hip/hip_bf16.h>
#include <math.h>

// ---------------------------------------------------------------------------
// FullQuantumModel on MI455X (gfx1250):
//   out[b] = || U_half @ s_b ||^2,  U_half = first 128 rows of the 256x256
//   circuit unitary. bf16x3-split WMMA GEMM (v_wmma_f32_16x16x32_bf16):
//   - S streamed from HBM exactly once (268 MB, ~11.5 us floor @ 23.3 TB/s),
//     with non-temporal loads so the stream doesn't evict packed U from L0/L2.
//   - U built on device (negligible) and packed into the exact wave32 32x16
//     bf16 B-matrix VGPR layout (4 matrices: Re/Im x hi/lo; the minus sign of
//     the complex product is folded into register-negated Im(S), saving 1/3
//     of B-side cache traffic vs storing -Im(U) tiles).
// ---------------------------------------------------------------------------

typedef __attribute__((ext_vector_type(16))) __bf16       v16bf;
typedef __attribute__((ext_vector_type(8)))  float        v8f;
typedef __attribute__((ext_vector_type(8)))  unsigned int v8u;
typedef __attribute__((ext_vector_type(4)))  float        f4;

#define NLAY 3
#define NQ   8
#define QDIM 256
#define HALF_DIM 128

struct c64 { float x, y; };
__device__ __forceinline__ c64 cmul(c64 a, c64 b) { return { a.x*b.x - a.y*b.y, a.x*b.y + a.y*b.x }; }
__device__ __forceinline__ c64 cadd(c64 a, c64 b) { return { a.x + b.x, a.y + b.y }; }

// round-to-nearest-even float -> bf16 bits
__device__ __forceinline__ unsigned int bfhi_bits(float x) {
    unsigned int u = __builtin_bit_cast(unsigned int, x);
    return (u + 0x7FFFu + ((u >> 16) & 1u)) >> 16;
}
__device__ __forceinline__ float bf_to_f(unsigned int b) {
    return __builtin_bit_cast(float, b << 16);
}
// split two consecutive f32 into packed bf16 hi/lo dwords (even K in [15:0])
struct HL { unsigned int h, l; };
__device__ __forceinline__ HL split2(float x0, float x1) {
    unsigned int h0 = bfhi_bits(x0), h1 = bfhi_bits(x1);
    unsigned int l0 = bfhi_bits(x0 - bf_to_f(h0));
    unsigned int l1 = bfhi_bits(x1 - bf_to_f(h1));
    HL r;
    r.h = h0 | (h1 << 16);
    r.l = l0 | (l1 << 16);
    return r;
}

// ---------------------------------------------------------------------------
// Kernel 1: simulate circuit on 256 basis vectors -> W[r*256 + t] = U[r][t].
// One block, 256 threads; thread t owns column t (columns independent, no
// barriers). Total work ~1.5M complex MACs: negligible.
// ---------------------------------------------------------------------------
__global__ void qsim_build_u(const float* __restrict__ params, c64* __restrict__ W) {
    const int t = threadIdx.x;            // column / input basis index
    for (int r = 0; r < QDIM; ++r)
        W[r * QDIM + t] = { (r == t) ? 1.0f : 0.0f, 0.0f };

    for (int l = 0; l < NLAY; ++l) {
        // single-qubit rotations U = RZ*RY*RX on wire q (bit b = 7-q, MSB=wire0)
        for (int q = 0; q < NQ; ++q) {
            const float* p = params + (l * NQ + q) * 3;
            float hx = 0.5f * p[0], hy = 0.5f * p[1], hz = 0.5f * p[2];
            float cx = cosf(hx), sx = sinf(hx);
            float cy = cosf(hy), sy = sinf(hy);
            float cz = cosf(hz), sz = sinf(hz);
            c64 m00{ cy * cx,  sy * sx }, m01{ -sy * cx, -cy * sx };
            c64 m10{ sy * cx, -cy * sx }, m11{  cy * cx, -sy * sx };
            c64 ezm{ cz, -sz }, ezp{ cz, sz };
            c64 u00 = cmul(ezm, m00), u01 = cmul(ezm, m01);
            c64 u10 = cmul(ezp, m10), u11 = cmul(ezp, m11);
            const int b = 7 - q, mask = 1 << b;
            for (int rl = 0; rl < HALF_DIM; ++rl) {
                int r0 = ((rl >> b) << (b + 1)) | (rl & (mask - 1));
                int r1 = r0 | mask;
                c64 a0 = W[r0 * QDIM + t];
                c64 a1 = W[r1 * QDIM + t];
                W[r0 * QDIM + t] = cadd(cmul(u00, a0), cmul(u01, a1));
                W[r1 * QDIM + t] = cadd(cmul(u10, a0), cmul(u11, a1));
            }
        }
        // ring of CNOTs: ctrl q -> tgt (q+1)%8, applied in order
        for (int q = 0; q < NQ; ++q) {
            const int c  = 7 - q;
            const int tb = 7 - ((q + 1) & 7);
            const int cm = 1 << c, tm = 1 << tb;
            for (int r = 0; r < QDIM; ++r) {
                if ((r & cm) && !(r & tm)) {
                    c64 a  = W[r * QDIM + t];
                    c64 bv = W[(r | tm) * QDIM + t];
                    W[r * QDIM + t] = bv;
                    W[(r | tm) * QDIM + t] = a;
                }
            }
        }
    }
}

// ---------------------------------------------------------------------------
// Kernel 2: pack B = U_half^T tiles into wave32 32x16 bf16 B-matrix layout.
// FOUR matrices per (kc, ct): {Re hi, Re lo, Im hi, Im lo} (sign of the
// complex product is folded into A). Flat layout:
//   Bp[(((kc*8+ct)*4+mat)*32 + lane)*8 + v]
// so a wave loads a B tile as two contiguous b128 per lane (1 KB per tile).
// B VGPR layout: lanes 0-15 hold K=0..7,16..23; lanes 16-31 hold K=8..15,24..31
// (dword v -> 2 consecutive K, even K in [15:0]). N = lane%16.
// ---------------------------------------------------------------------------
__global__ void qpack_b(const c64* __restrict__ W, unsigned int* __restrict__ Bp) {
    const int idx  = blockIdx.x * blockDim.x + threadIdx.x;  // < 8*8*4*256
    const int v    = idx & 7;
    const int lane = (idx >> 3) & 31;
    const int rem  = idx >> 8;                 // (kc*8 + ct)*4 + mat
    const int mat  = rem & 3;
    const int ct   = (rem >> 2) & 7;
    const int kc   = rem >> 5;
    const int n    = ct * 16 + (lane & 15);    // output row of U_half (< 128)
    const int kb   = kc * 32 + ((lane >> 4) << 3) + 2 * v + ((v >> 2) << 3);

    unsigned int d = 0;
    #pragma unroll
    for (int e = 0; e < 2; ++e) {
        c64 u = W[n * QDIM + kb + e];
        float x = (mat < 2) ? u.x : u.y;
        unsigned int hb = bfhi_bits(x);
        if (mat & 1) hb = bfhi_bits(x - bf_to_f(hb));   // lo part
        d |= hb << (16 * e);
    }
    Bp[idx] = d;
}

// ---------------------------------------------------------------------------
// Kernel 3: main GEMM + |.|^2 reduction. One wave per 16 batch rows.
// K-outer loop (8 chunks of 32), all 8 column tiles accumulated in registers
// (16 v8f accumulators = 128 VGPRs) so S streams from HBM exactly once.
// ---------------------------------------------------------------------------
__device__ __forceinline__ v16bf ldb(const unsigned int* p) {
    return __builtin_bit_cast(v16bf, *(const v8u*)p);
}
__device__ __forceinline__ v8f wmma_bf(v16bf a, v16bf b, v8f c) {
    return __builtin_amdgcn_wmma_f32_16x16x32_bf16(false, a, false, b, (short)0, c, false, false);
}

__global__ __launch_bounds__(256) void qgemm(const float* __restrict__ Sre,
                                             const float* __restrict__ Sim,
                                             const unsigned int* __restrict__ Bp,
                                             float* __restrict__ out, int batch) {
    const int lane = threadIdx.x & 31;
    const int wave = blockIdx.x * (blockDim.x >> 5) + (threadIdx.x >> 5);
    const int row0 = wave * 16;
    if (row0 >= batch) return;                 // wave-uniform: EXEC stays all-1s

    const int m  = lane & 15;                  // A-matrix row (M)
    const int hf = lane >> 4;                  // lane half selects K groups
    const int base0 = hf * 8;                  // A elems 0..7  -> K = base0..base0+7
    const int base1 = 16 + hf * 8;             // A elems 8..15 -> K = base1..base1+7

    const float* pre = Sre + (size_t)(row0 + m) * QDIM;
    const float* pim = Sim + (size_t)(row0 + m) * QDIM;

    v8f accR[8], accI[8];
    #pragma unroll
    for (int ct = 0; ct < 8; ++ct) { accR[ct] = (v8f)0.0f; accI[ct] = (v8f)0.0f; }

    for (int kc = 0; kc < 8; ++kc) {
        if (kc < 7) {                          // gfx1250 global_prefetch for next chunk
            __builtin_prefetch(pre + (kc + 1) * 32 + base0, 0, 0);
            __builtin_prefetch(pim + (kc + 1) * 32 + base0, 0, 0);
        }
        // ---- stream S with NT loads (read-once; keep packed U cached),
        //      split into bf16 hi/lo
        f4 fr0 = __builtin_nontemporal_load((const f4*)(pre + kc * 32 + base0));
        f4 fr1 = __builtin_nontemporal_load((const f4*)(pre + kc * 32 + base0) + 1);
        f4 fr2 = __builtin_nontemporal_load((const f4*)(pre + kc * 32 + base1));
        f4 fr3 = __builtin_nontemporal_load((const f4*)(pre + kc * 32 + base1) + 1);
        f4 fi0 = __builtin_nontemporal_load((const f4*)(pim + kc * 32 + base0));
        f4 fi1 = __builtin_nontemporal_load((const f4*)(pim + kc * 32 + base0) + 1);
        f4 fi2 = __builtin_nontemporal_load((const f4*)(pim + kc * 32 + base1));
        f4 fi3 = __builtin_nontemporal_load((const f4*)(pim + kc * 32 + base1) + 1);

        float xr[16] = { fr0.x, fr0.y, fr0.z, fr0.w, fr1.x, fr1.y, fr1.z, fr1.w,
                         fr2.x, fr2.y, fr2.z, fr2.w, fr3.x, fr3.y, fr3.z, fr3.w };
        float xi[16] = { fi0.x, fi0.y, fi0.z, fi0.w, fi1.x, fi1.y, fi1.z, fi1.w,
                         fi2.x, fi2.y, fi2.z, fi2.w, fi3.x, fi3.y, fi3.z, fi3.w };

        v8u hR, lR, hI, lI, nhI, nlI;
        #pragma unroll
        for (int j = 0; j < 8; ++j) {
            HL a = split2(xr[2 * j], xr[2 * j + 1]);
            hR[j] = a.h; lR[j] = a.l;
            HL b = split2(xi[2 * j], xi[2 * j + 1]);
            hI[j] = b.h; lI[j] = b.l;
            nhI[j] = b.h ^ 0x80008000u;        // -Im(S): packed bf16 sign flip
            nlI[j] = b.l ^ 0x80008000u;
        }

        v16bf aRh  = __builtin_bit_cast(v16bf, hR);
        v16bf aRl  = __builtin_bit_cast(v16bf, lR);
        v16bf aIh  = __builtin_bit_cast(v16bf, hI);
        v16bf aIl  = __builtin_bit_cast(v16bf, lI);
        v16bf aInh = __builtin_bit_cast(v16bf, nhI);
        v16bf aInl = __builtin_bit_cast(v16bf, nlI);

        const unsigned int* bbase = Bp + (size_t)kc * 8 * 4 * 256 + lane * 8;
        #pragma unroll
        for (int ct = 0; ct < 8; ++ct) {
            const unsigned int* bp = bbase + ct * 4 * 256;
            v16bf brh = ldb(bp);              // Re(U) hi
            v16bf brl = ldb(bp + 256);        // Re(U) lo
            v16bf bih = ldb(bp + 512);        // Im(U) hi
            v16bf bil = ldb(bp + 768);        // Im(U) lo
            // P_re += Sre*Ur + (-Sim)*Ui   (bf16x3 split each product)
            accR[ct] = wmma_bf(aRh,  brh, accR[ct]);
            accR[ct] = wmma_bf(aRl,  brh, accR[ct]);
            accR[ct] = wmma_bf(aRh,  brl, accR[ct]);
            accR[ct] = wmma_bf(aInh, bih, accR[ct]);
            accR[ct] = wmma_bf(aInl, bih, accR[ct]);
            accR[ct] = wmma_bf(aInh, bil, accR[ct]);
            // P_im += Sre*Ui + Sim*Ur
            accI[ct] = wmma_bf(aRh,  bih, accI[ct]);
            accI[ct] = wmma_bf(aRl,  bih, accI[ct]);
            accI[ct] = wmma_bf(aRh,  bil, accI[ct]);
            accI[ct] = wmma_bf(aIh,  brh, accI[ct]);
            accI[ct] = wmma_bf(aIl,  brh, accI[ct]);
            accI[ct] = wmma_bf(aIh,  brl, accI[ct]);
        }
    }

    // ---- epilogue: out[row] = sum_n (P_re^2 + P_im^2)
    float s[8];
    #pragma unroll
    for (int v = 0; v < 8; ++v) s[v] = 0.0f;
    #pragma unroll
    for (int ct = 0; ct < 8; ++ct) {
        #pragma unroll
        for (int v = 0; v < 8; ++v)
            s[v] += accR[ct][v] * accR[ct][v] + accI[ct][v] * accI[ct][v];
    }
    // reduce over N (16 lanes per half; xor masks stay within each half)
    #pragma unroll
    for (int mask = 1; mask < 16; mask <<= 1) {
        #pragma unroll
        for (int v = 0; v < 8; ++v) s[v] += __shfl_xor(s[v], mask, 32);
    }
    if ((lane & 15) == 0) {
        #pragma unroll
        for (int v = 0; v < 8; ++v) out[row0 + hf * 8 + v] = s[v];
    }
}

// ---------------------------------------------------------------------------
extern "C" void kernel_launch(void* const* d_in, const int* in_sizes, int n_in,
                              void* d_out, int out_size, void* d_ws, size_t ws_size,
                              hipStream_t stream) {
    const float* params = (const float*)d_in[0];
    const float* sre    = (const float*)d_in[1];
    const float* sim    = (const float*)d_in[2];
    float* out          = (float*)d_out;
    const int batch     = in_sizes[1] / QDIM;

    c64* W = (c64*)d_ws;                                     // 256*256*8 = 512 KB
    unsigned int* Bp = (unsigned int*)((char*)d_ws + (size_t)QDIM * QDIM * sizeof(c64)); // 256 KB

    qsim_build_u<<<1, 256, 0, stream>>>(params, W);
    qpack_b<<<(8 * 8 * 4 * 256) / 256, 256, 0, stream>>>(W, Bp);

    const int waves  = batch / 16;
    const int wpb    = 256 / 32;
    const int blocks = (waves + wpb - 1) / wpb;
    qgemm<<<blocks, 256, 0, stream>>>(sre, sim, Bp, out, batch);
}
